// VariationalLSTM_44367012168009
// MI455X (gfx1250) — matrix operand order repaired and checked by reading, not compiled, not run on hardware
//
#include <hip/hip_runtime.h>
#include <hip/hip_bf16.h>

typedef __attribute__((ext_vector_type(16))) _Float16 v16h;
typedef __attribute__((ext_vector_type(8)))  float    v8f;

#define Bn   32
#define Tn   512
#define Hn   400
#define G4   1600          // 4*H
#define NTl  100           // G4/16 n-tiles
#define KT_H 13            // ceil(400/32) k-tiles for K=400 (padded to 416)
#define KT_2H 25           // 800/32 k-tiles for K=800
#define BT   (Bn*Tn)       // 16384 rows
#define MTl  (BT/16)       // 1024 m-tiles

__device__ __forceinline__ float sigmf(float x){ return 1.f/(1.f+__expf(-x)); }

// 16-bit WMMA fragment K map: half-index i + lane-half -> K within 32
__device__ __forceinline__ int kmap16(int i, int laneHi){
    return ((i < 8) ? i : (i + 8)) + 8*laneHi;
}

// ---------------------------------------------------------------- fill
__global__ void fill_zero(float* __restrict__ p, int n){
    for (int i = blockIdx.x*blockDim.x + threadIdx.x; i < n; i += gridDim.x*blockDim.x)
        p[i] = 0.f;
}

// ---------------------------------------------------------------- weight pack (B-fragment major)
// src: row-major [1600, Ksrc] f32  ->  dst: [ntile][ktile][lane][16] f16
__global__ void pack_bfrag(const float* __restrict__ src, _Float16* __restrict__ dst,
                           int Ksrc, int kTiles, int totalHalves){
    int idx = blockIdx.x*blockDim.x + threadIdx.x;
    if (idx >= totalHalves) return;
    int i     = idx & 15;
    int lane  = (idx >> 4) & 31;
    int tile  = idx >> 9;              // ntile*kTiles + kt
    int kt    = tile % kTiles;
    int ntile = tile / kTiles;
    int N = ntile*16 + (lane & 15);
    int K = kt*32 + kmap16(i, lane >> 4);
    float v = (K < Ksrc) ? src[(size_t)N*Ksrc + K] : 0.f;
    dst[idx] = (_Float16)v;
}

// ---------------------------------------------------------------- activation pack (A-fragment major)
// src: row-major [BT, Ksrc] f32 -> dst: [mtile][ktile][lane][16] f16 (zero-padded K)
__global__ void pack_afrag(const float* __restrict__ src, _Float16* __restrict__ dst,
                           int Ksrc, int kTiles, int totalHalves){
    int idx = blockIdx.x*blockDim.x + threadIdx.x;
    if (idx >= totalHalves) return;
    int i     = idx & 15;
    int lane  = (idx >> 4) & 31;
    int tile  = idx >> 9;              // mtile*kTiles + kt
    int kt    = tile % kTiles;
    int mtile = tile / kTiles;
    int row = mtile*16 + (lane & 15);  // A-frag: M = lane%16
    int K   = kt*32 + kmap16(i, lane >> 4);
    float v = (K < Ksrc) ? src[(size_t)row*Ksrc + K] : 0.f;
    dst[idx] = (_Float16)v;
}

// ---------------------------------------------------------------- input-gate GEMM
// Packed-A x Packed-B; XG stored gate-interleaved: XG[dir][row][h][g], g in {i,f,g,o}
__global__ void gemm_xg(const _Float16* __restrict__ Xp, int kTiles,
                        const _Float16* __restrict__ Wp, const float* __restrict__ bias,
                        float* __restrict__ XG){
    int wave = (blockIdx.x*blockDim.x + threadIdx.x) >> 5;
    int lane = threadIdx.x & 31;
    int dir   = wave / (MTl*NTl);
    int rem   = wave % (MTl*NTl);
    int mtile = rem / NTl;
    int ntile = rem % NTl;
    int laneHi = lane >> 4;
    int col    = lane & 15;
    const _Float16* wdir = Wp + (size_t)dir * NTl * kTiles * 512;
    const _Float16* xmt  = Xp + ((size_t)mtile * kTiles << 9);

    v8f acc = {};
    for (int kt = 0; kt < kTiles; ++kt){
        v16h a = *(const v16h*)(xmt + ((size_t)kt << 9) + (lane << 4));
        v16h b = *(const v16h*)(wdir + ((size_t)(ntile*kTiles + kt) << 9) + (lane << 4));
        acc = __builtin_amdgcn_wmma_f32_16x16x32_f16(false, a, false, b, (short)0, acc, false, false);
    }
    int gsel = ntile / 25;                 // gate index (400 = 25 n-tiles per gate)
    int hh   = (ntile % 25)*16 + col;      // h index within gate
    float bv = bias[dir*G4 + gsel*Hn + hh];
    float* out = XG + (size_t)dir * BT * G4;
    #pragma unroll
    for (int r = 0; r < 8; ++r){
        int row = mtile*16 + r + 8*laneHi; // D-frag: M = r + 8*(lane>=16)
        out[((size_t)row * Hn + hh)*4 + gsel] = acc[r] + bv;
    }
}

// ---------------------------------------------------------------- recurrent scan
// grid.x = 2 (dir), 800 threads = 25 waves; wave w owns h-cols [16w,16w+16)
__global__ void __launch_bounds__(800, 1)
lstm_scan(const float* __restrict__ XG, const _Float16* __restrict__ WhhP,
          const int* __restrict__ lengths,
          _Float16* __restrict__ outP,    // next-layer packed A-frag input (kTiles=25) or null
          float* __restrict__ outF,       // final f32 sequence output [B][T][800] or null
          float* __restrict__ hn)         // d_out h_n slot base for this layer
{
    __shared__ __attribute__((aligned(32))) _Float16 ldsA[KT_H*2*512]; // h A-frags, 26.6 KB
    __shared__ int llen[Bn];

    const int dir    = blockIdx.x;
    const int tid    = threadIdx.x;
    const int ht     = tid >> 5;          // wave id = h-tile 0..24
    const int lane   = tid & 31;
    const int col    = lane & 15;
    const int laneHi = lane >> 4;

    for (int i = tid; i < KT_H*2*512; i += blockDim.x) ldsA[i] = (_Float16)0.f;
    if (tid < Bn) llen[tid] = lengths[tid];
    __syncthreads();

    const float4*   xg4 = (const float4*)(XG + (size_t)dir * BT * G4);
    const _Float16* whh = WhhP + (size_t)dir * NTl * KT_H * 512;

    float cst[2][8], hlast[2][8];
    #pragma unroll
    for (int mt = 0; mt < 2; ++mt)
        #pragma unroll
        for (int r = 0; r < 8; ++r){ cst[mt][r] = 0.f; hlast[mt][r] = 0.f; }

    // LDS h scatter constants: element (M=b_local, K=(ht&1)*16+col)
    const int kt_w    = ht >> 1;
    const int laneSel = (col >= 8) ? 1 : 0;
    const int Kp      = ((ht & 1) << 4) + col - 8*laneSel;
    const int ihalf   = (Kp < 8) ? Kp : (Kp - 8);
    // packed-output scatter constants: k = dir*400 + ht*16 + col, kTiles = 25
    const int eP      = dir*25 + ht;
    const int ktP     = eP >> 1;
    const int KpP     = ((eP & 1) << 4) + col - 8*laneSel;
    const int ihalfP  = (KpP < 8) ? KpP : (KpP - 8);

    for (int s = 0; s < Tn; ++s){
        v8f acc[4][2];
        // init accumulators from gate-interleaved XG (one b128 per (mt,r))
        #pragma unroll
        for (int mt = 0; mt < 2; ++mt){
            #pragma unroll
            for (int r = 0; r < 8; ++r){
                int b   = mt*16 + r + 8*laneHi;
                int len = llen[b];
                int t   = dir ? (len - 1 - s) : s;
                t = (t < 0) ? 0 : t;
                float4 g4 = xg4[(size_t)(b*Tn + t) * Hn + ht*16 + col];
                acc[0][mt][r] = g4.x; acc[1][mt][r] = g4.y;
                acc[2][mt][r] = g4.z; acc[3][mt][r] = g4.w;
            }
        }
        // gates += h @ Whh^T  (A from LDS, B from L2-resident packed weights)
        for (int kt = 0; kt < KT_H; ++kt){
            v16h a0 = *(const v16h*)&ldsA[(kt*2 + 0)*512 + (lane << 4)];
            v16h a1 = *(const v16h*)&ldsA[(kt*2 + 1)*512 + (lane << 4)];
            #pragma unroll
            for (int g = 0; g < 4; ++g){
                v16h bf = *(const v16h*)(whh + ((size_t)((g*25 + ht)*KT_H + kt) << 9) + (lane << 4));
                acc[g][0] = __builtin_amdgcn_wmma_f32_16x16x32_f16(false, a0, false, bf, (short)0, acc[g][0], false, false);
                acc[g][1] = __builtin_amdgcn_wmma_f32_16x16x32_f16(false, a1, false, bf, (short)0, acc[g][1], false, false);
            }
        }
        __syncthreads();   // all A reads done before anyone rewrites h
        // elementwise LSTM cell; c lives in registers, masked-hold semantics
        #pragma unroll
        for (int mt = 0; mt < 2; ++mt){
            #pragma unroll
            for (int r = 0; r < 8; ++r){
                int b   = mt*16 + r + 8*laneHi;
                int len = llen[b];
                bool m  = (s < len);
                int t   = dir ? (len - 1 - s) : s;
                float gi = acc[0][mt][r], gf = acc[1][mt][r];
                float gg = acc[2][mt][r], go = acc[3][mt][r];
                float c2 = sigmf(gf)*cst[mt][r] + sigmf(gi)*tanhf(gg);
                float h2 = sigmf(go)*tanhf(c2);
                if (m){
                    cst[mt][r]   = c2;
                    hlast[mt][r] = h2;
                    if (outF)
                        outF[((size_t)b * Tn + t) * (2*Hn) + dir*Hn + ht*16 + col] = h2;
                    if (outP){
                        int row = b*Tn + t;
                        size_t pos = ((size_t)((row >> 4)*KT_2H + ktP) << 9)
                                   + (((row & 15) + 16*laneSel) << 4) + ihalfP;
                        outP[pos] = (_Float16)h2;
                    }
                    int lanep = (r + 8*laneHi) + 16*laneSel;
                    ldsA[((kt_w*2 + mt) << 9) + (lanep << 4) + ihalf] = (_Float16)h2;
                }
            }
        }
        __syncthreads();   // new h visible before next step's A reads
    }
    // final hidden state -> h_n and c_n (reference returns h for both)
    #pragma unroll
    for (int mt = 0; mt < 2; ++mt)
        #pragma unroll
        for (int r = 0; r < 8; ++r){
            int b = mt*16 + r + 8*laneHi;
            size_t o = (size_t)dir*(Bn*Hn) + (size_t)b*Hn + ht*16 + col;
            hn[o]           = hlast[mt][r];
            hn[o + 6*Bn*Hn] = hlast[mt][r];
        }
}

// ---------------------------------------------------------------- host
extern "C" void kernel_launch(void* const* d_in, const int* in_sizes, int n_in,
                              void* d_out, int out_size, void* d_ws, size_t ws_size,
                              hipStream_t stream){
    const float* x     = (const float*)d_in[0];
    const int*   lens  = (const int*)d_in[1];
    const float* w_ih0 = (const float*)d_in[2];   // [2,1600,400]
    const float* w_hh0 = (const float*)d_in[3];   // [2,1600,400]
    const float* b0    = (const float*)d_in[4];   // [2,1600]
    const float* w_ih  = (const float*)d_in[5];   // [2,2,1600,800]
    const float* w_hh  = (const float*)d_in[6];   // [2,2,1600,400]
    const float* bb    = (const float*)d_in[7];   // [2,2,1600]
    float* out = (float*)d_out;

    const size_t FR_H  = (size_t)NTl*KT_H*512;    // 665,600 halves  (B-frags, K=400)
    const size_t FR_2H = (size_t)NTl*KT_2H*512;   // 1,280,000 halves (B-frags, K=800)
    const size_t AX_H  = (size_t)MTl*KT_H*512;    // 6,815,744 halves (A-frags, K=400)
    const size_t AX_2H = (size_t)MTl*KT_2H*512;   // 13,107,200 halves (A-frags, K=800)

    char* ws = (char*)d_ws;
    size_t off = 0;
    auto alloc = [&](size_t bytes)->void*{
        void* p = ws + off;
        off = (off + bytes + 255) & ~(size_t)255;
        return p;
    };
    _Float16* pWih0  = (_Float16*)alloc(2*FR_H  * 2);
    _Float16* pWih12 = (_Float16*)alloc(4*FR_2H * 2);
    _Float16* pWhh   = (_Float16*)alloc(6*FR_H  * 2);
    _Float16* pX0    = (_Float16*)alloc(AX_H    * 2);  // packed layer-0 input
    _Float16* pA1    = (_Float16*)alloc(AX_2H   * 2);  // packed layer-1 input (layer-0 out)
    _Float16* pA2    = (_Float16*)alloc(AX_2H   * 2);  // packed layer-2 input (layer-1 out)
    float* xg = (float*)alloc((size_t)2*BT*G4 * 4);

    const size_t XTOT = (size_t)BT*800;           // sequence-output floats in d_out

    // zero final output + packed inter-layer buffers (padded positions must be 0)
    fill_zero<<<4096, 256, 0, stream>>>(out, out_size);
    fill_zero<<<4096, 256, 0, stream>>>((float*)pA1, (int)(AX_2H/2));
    fill_zero<<<4096, 256, 0, stream>>>((float*)pA2, (int)(AX_2H/2));

    // pack weights (B-frag) and layer-0 activations (A-frag)
    for (int dir = 0; dir < 2; ++dir){
        pack_bfrag<<<(int)((FR_H+255)/256), 256, 0, stream>>>(
            w_ih0 + (size_t)dir*1600*400, pWih0 + dir*FR_H, 400, KT_H, (int)FR_H);
        pack_bfrag<<<(int)((FR_H+255)/256), 256, 0, stream>>>(
            w_hh0 + (size_t)dir*1600*400, pWhh + dir*FR_H, 400, KT_H, (int)FR_H);
        for (int l = 1; l < 3; ++l){
            pack_bfrag<<<(int)((FR_2H+255)/256), 256, 0, stream>>>(
                w_ih + (size_t)((l-1)*2 + dir)*1600*800, pWih12 + ((l-1)*2 + dir)*FR_2H,
                800, KT_2H, (int)FR_2H);
            pack_bfrag<<<(int)((FR_H+255)/256), 256, 0, stream>>>(
                w_hh + (size_t)((l-1)*2 + dir)*1600*400, pWhh + (l*2 + dir)*FR_H,
                400, KT_H, (int)FR_H);
        }
    }
    pack_afrag<<<(int)((AX_H+255)/256), 256, 0, stream>>>(x, pX0, 400, KT_H, (int)AX_H);

    const int gemmBlocks = (2*MTl*NTl) / 8;       // 25,600 blocks of 8 waves

    // layer 0
    gemm_xg<<<gemmBlocks, 256, 0, stream>>>(pX0, KT_H, pWih0, b0, xg);
    lstm_scan<<<2, 800, 0, stream>>>(xg, pWhh,          lens, pA1, nullptr,
                                     out + XTOT + 0*2*Bn*Hn);
    // layer 1
    gemm_xg<<<gemmBlocks, 256, 0, stream>>>(pA1, KT_2H, pWih12, bb, xg);
    lstm_scan<<<2, 800, 0, stream>>>(xg, pWhh + 2*FR_H, lens, pA2, nullptr,
                                     out + XTOT + 1*2*Bn*Hn);
    // layer 2 (sequence output straight to d_out)
    gemm_xg<<<gemmBlocks, 256, 0, stream>>>(pA2, KT_2H, pWih12 + 2*FR_2H, bb + 3200, xg);
    lstm_scan<<<2, 800, 0, stream>>>(xg, pWhh + 4*FR_H, lens, nullptr, out,
                                     out + XTOT + 2*2*Bn*Hn);
}
